// GAT_45389214384225
// MI455X (gfx1250) — compile-verified, matrix operand order checked by reading, not run on hardware
//
#include <hip/hip_runtime.h>
#include <hip/hip_bf16.h>

// Problem constants (from reference)
#define NN   4096
#define TT   16
#define DIN  128
#define HH   8
#define CC   64
#define KOUT 512              // H*C
#define MTOT (NN*TT)          // 65536
#define NEGV (-9.0e15f)
#define ALPHA 0.01f

typedef __attribute__((ext_vector_type(16))) __bf16 v16bf;
typedef __attribute__((ext_vector_type(8)))  float  v8f;

union BfVec { v16bf v; uint4 q[2]; unsigned short u[16]; };

__device__ __forceinline__ unsigned short f2bf(float f) {
    union { float f; unsigned int u; } x; x.f = f;
    unsigned int r = x.u + 0x7FFFu + ((x.u >> 16) & 1u);   // RNE
    return (unsigned short)(r >> 16);
}
__device__ __forceinline__ float bf2f(unsigned short b) {
    union { unsigned int u; float f; } x; x.u = ((unsigned int)b) << 16;
    return x.f;
}
__device__ __forceinline__ unsigned int pk2bf(float a, float b) {
    return (unsigned int)f2bf(a) | ((unsigned int)f2bf(b) << 16);
}
__device__ __forceinline__ float lrelu(float z) {   // branchless, ALPHA < 1
    return fmaxf(z, ALPHA * z);
}

// 16-bit WMMA fragment: lane L (n = L&15, kh = L>>4) holds K runs
// {kh*8 .. kh*8+7} and {16+kh*8 .. 16+kh*8+7}  -> two 16B contiguous chunks
// when the LDS tile is stored K-contiguous with row stride 40 ushorts (80B).

// ---------------------------------------------------------------------------
// Kernel 1: h = x @ fc_w^T + fc_b   (M=65536, K=128, N=512), store bf16
// ---------------------------------------------------------------------------
__global__ __launch_bounds__(256) void fc_kernel(
    const float* __restrict__ x, const float* __restrict__ fc_w,
    const float* __restrict__ fc_b, unsigned short* __restrict__ hbf)
{
    const int bid   = blockIdx.x;
    const int mtile = bid >> 2;           // 0..4095
    const int ctile = bid & 3;            // 0..3
    const int m0    = mtile * 16;
    const int c0    = ctile * 128;
    const int tid   = threadIdx.x;
    const int wv    = tid >> 5;
    const int ln    = tid & 31;
    const int n     = ln & 15;
    const int kh    = ln >> 4;

    __shared__ __align__(16) unsigned short xs[16 * 136];   // [r][d] stride 136
    __shared__ __align__(16) unsigned short ws[128 * 136];  // [col][d] stride 136

    for (int idx = tid; idx < 16 * 32; idx += 256) {
        int r = idx >> 5, d4 = (idx & 31) * 4;
        float4 f = *(const float4*)&x[(m0 + r) * DIN + d4];
        uint2 o; o.x = pk2bf(f.x, f.y); o.y = pk2bf(f.z, f.w);
        *(uint2*)&xs[r * 136 + d4] = o;
    }
    for (int idx = tid; idx < 128 * 32; idx += 256) {
        int col = idx >> 5, d4 = (idx & 31) * 4;
        float4 f = *(const float4*)&fc_w[(c0 + col) * DIN + d4];
        uint2 o; o.x = pk2bf(f.x, f.y); o.y = pk2bf(f.z, f.w);
        *(uint2*)&ws[col * 136 + d4] = o;
    }
    __syncthreads();

    const int colbase = wv * 16;
    v8f acc = {};
#pragma unroll
    for (int ks = 0; ks < 4; ++ks) {
        BfVec A, B;
        A.q[0] = *(const uint4*)&xs[n * 136 + ks * 32 + kh * 8];
        A.q[1] = *(const uint4*)&xs[n * 136 + ks * 32 + 16 + kh * 8];
        B.q[0] = *(const uint4*)&ws[(colbase + n) * 136 + ks * 32 + kh * 8];
        B.q[1] = *(const uint4*)&ws[(colbase + n) * 136 + ks * 32 + 16 + kh * 8];
        acc = __builtin_amdgcn_wmma_f32_16x16x32_bf16(
            false, A.v, false, B.v, (short)0, acc, false, false);
    }
#pragma unroll
    for (int r = 0; r < 8; ++r) {
        int m   = m0 + r + 8 * kh;                    // D layout: lanes>=16 -> M+8
        int col = c0 + colbase + n;
        hbf[m * KOUT + col] = f2bf(acc[r] + fc_b[col]);
    }
}

// ---------------------------------------------------------------------------
// Kernel 2: fused time-aggregation + attention projections
// ---------------------------------------------------------------------------
__global__ __launch_bounds__(256) void srcdst_kernel(
    const unsigned short* __restrict__ hbf, const float* __restrict__ ta_w,
    const float* __restrict__ ta_b, const float* __restrict__ a,
    float* __restrict__ src, float* __restrict__ dst)
{
    int t0 = blockIdx.x * blockDim.x + threadIdx.x;   // (n*8 + h)
    if (t0 >= NN * HH) return;
    int nidx = t0 >> 3, h = t0 & 7;
    float tw[TT];
#pragma unroll
    for (int t = 0; t < TT; ++t) tw[t] = ta_w[t];
    float tb = ta_b[0];
    float s = 0.f, d = 0.f;
    for (int c = 0; c < CC; ++c) {
        float xf = tb;
#pragma unroll
        for (int t = 0; t < TT; ++t)
            xf += bf2f(hbf[(nidx * TT + t) * KOUT + h * CC + c]) * tw[t];
        s += xf * a[h * 2 * CC + c];
        d += xf * a[h * 2 * CC + CC + c];
    }
    src[t0] = s;
    dst[t0] = d;
}

// ---------------------------------------------------------------------------
// Kernel 3: two-pass masked-softmax stats per (row i, all 8 heads)
// pass 1: block max; pass 2: sum of exp  -> 1 transcendental per element
// ---------------------------------------------------------------------------
__global__ __launch_bounds__(256) void stats_kernel(
    const int* __restrict__ adj, const float* __restrict__ src,
    const float* __restrict__ dst, float* __restrict__ rowmax,
    float* __restrict__ rowsum)
{
    const int i = blockIdx.x;
    const int tid = threadIdx.x;
    __shared__ float red[HH][256];
    __shared__ float mx[HH];

    float si[HH];
#pragma unroll
    for (int h = 0; h < HH; ++h) si[h] = src[i * HH + h];

    // ---- pass 1: masked max ----
    float m[HH];
#pragma unroll
    for (int h = 0; h < HH; ++h) m[h] = -3.0e38f;
    for (int j = tid; j < NN; j += 256) {
        int aij = adj[i * NN + j];
#pragma unroll
        for (int h = 0; h < HH; ++h) {
            float v = (aij != 0) ? lrelu(si[h] + dst[j * HH + h]) : NEGV;
            m[h] = fmaxf(m[h], v);
        }
    }
#pragma unroll
    for (int h = 0; h < HH; ++h) red[h][tid] = m[h];
    __syncthreads();
    for (int st = 128; st > 0; st >>= 1) {
        if (tid < st) {
#pragma unroll
            for (int h = 0; h < HH; ++h)
                red[h][tid] = fmaxf(red[h][tid], red[h][tid + st]);
        }
        __syncthreads();
    }
    if (tid < HH) { mx[tid] = red[tid][0]; rowmax[i * HH + tid] = red[tid][0]; }
    __syncthreads();
    float mloc[HH];
#pragma unroll
    for (int h = 0; h < HH; ++h) mloc[h] = mx[h];

    // ---- pass 2: sum of exp (adj row is cache-resident) ----
    float s[HH];
#pragma unroll
    for (int h = 0; h < HH; ++h) s[h] = 0.f;
    for (int j = tid; j < NN; j += 256) {
        int aij = adj[i * NN + j];
#pragma unroll
        for (int h = 0; h < HH; ++h) {
            float v = (aij != 0) ? lrelu(si[h] + dst[j * HH + h]) : NEGV;
            s[h] += __expf(v - mloc[h]);
        }
    }
#pragma unroll
    for (int h = 0; h < HH; ++h) red[h][tid] = s[h];
    __syncthreads();
    for (int st = 128; st > 0; st >>= 1) {
        if (tid < st) {
#pragma unroll
            for (int h = 0; h < HH; ++h)
                red[h][tid] += red[h][tid + st];
        }
        __syncthreads();
    }
    if (tid < HH) rowsum[i * HH + tid] = red[tid][0];
}

// ---------------------------------------------------------------------------
// Kernel 4: out[i,t,c] = (1/H) * sum_h sum_j p[i,j,h] * h[j,t,h,c]
// Double-buffered LDS; pA[row][k], Bs[col][k] (v_perm_b32 4x4 transpose on
// staging) -> every fragment is 2 x ds_load_b128; acc over j AND heads.
// ---------------------------------------------------------------------------
__global__ __launch_bounds__(256) void agg_kernel(
    const unsigned short* __restrict__ hbf, const int* __restrict__ adj,
    const float* __restrict__ src, const float* __restrict__ dst,
    const float* __restrict__ rowmax, const float* __restrict__ rowsum,
    float* __restrict__ out)
{
    const int itile = blockIdx.x;          // 0..255
    const int cc    = blockIdx.y;          // 0..1 column chunk (512 cols each)
    const int i0    = itile * 16;
    const int cbase = cc * 512;
    const int tid   = threadIdx.x;
    const int wv    = tid >> 5;
    const int ln    = tid & 31;
    const int n     = ln & 15;
    const int kh    = ln >> 4;

    __shared__ __align__(16) unsigned short pA[2][16 * 40];    // [row][k] pad 40
    __shared__ __align__(16) unsigned short Bs[2][512 * 40];   // [col][k] pad 40

    v8f acc[4] = {};

    auto stage = [&](int step, int buf) {
        const int h  = step >> 7;             // 0..7
        const int j0 = (step & 127) << 5;     // 0..4064
        // --- 16x32 probability tile (bf16), K-contiguous rows ---
        for (int idx = tid; idx < 512; idx += 256) {
            int r = idx >> 5, k = idx & 31;
            int i = i0 + r, j = j0 + k;
            float v = (adj[i * NN + j] != 0)
                          ? lrelu(src[i * HH + h] + dst[j * HH + h])
                          : NEGV;
            float p = __expf(v - rowmax[i * HH + h]) *
                      __builtin_amdgcn_rcpf(rowsum[i * HH + h]);
            pA[buf][r * 40 + k] = f2bf(p);
        }
        // --- B tile: 4x4 v_perm_b32 transpose blocks, coalesced reads ---
        for (int b = tid; b < 1024; b += 256) {
            int k0  = (b >> 7) * 4;           // 0,4,..,28
            int c0L = (b & 127) * 4;          // 0,4,..,508
            int colG = cbase + c0L;
            int t = colG >> 6, c = colG & 63;
            const unsigned short* gp = &hbf[((j0 + k0) * TT + t) * KOUT + h * CC + c];
            uint2 r0 = *(const uint2*)(gp);
            uint2 r1 = *(const uint2*)(gp + 8192);     // next j row: 16*512
            uint2 r2 = *(const uint2*)(gp + 16384);
            uint2 r3 = *(const uint2*)(gp + 24576);
#pragma unroll
            for (int dc = 0; dc < 4; ++dc) {
                unsigned int sel = (dc & 1) ? 0x07060302u : 0x05040100u;
                unsigned int a0 = (dc >> 1) ? r0.y : r0.x;
                unsigned int a1 = (dc >> 1) ? r1.y : r1.x;
                unsigned int a2 = (dc >> 1) ? r2.y : r2.x;
                unsigned int a3 = (dc >> 1) ? r3.y : r3.x;
                uint2 o;
                o.x = __builtin_amdgcn_perm(a1, a0, sel);   // rows k0,k0+1
                o.y = __builtin_amdgcn_perm(a3, a2, sel);   // rows k0+2,k0+3
                *(uint2*)&Bs[buf][(c0L + dc) * 40 + k0] = o;  // ds_store_b64
            }
        }
    };

    const int TOTAL = HH * (NN / 32);   // 1024 steps
    stage(0, 0);
    __syncthreads();
    for (int step = 0; step < TOTAL; ++step) {
        const int cur = step & 1;
        if (step + 1 < TOTAL) stage(step + 1, cur ^ 1);   // prefetch next tile
        BfVec A;
        A.q[0] = *(const uint4*)&pA[cur][n * 40 + kh * 8];
        A.q[1] = *(const uint4*)&pA[cur][n * 40 + 16 + kh * 8];
#pragma unroll
        for (int s = 0; s < 4; ++s) {
            int colL = wv * 64 + s * 16 + n;
            BfVec B;
            B.q[0] = *(const uint4*)&Bs[cur][colL * 40 + kh * 8];
            B.q[1] = *(const uint4*)&Bs[cur][colL * 40 + 16 + kh * 8];
            acc[s] = __builtin_amdgcn_wmma_f32_16x16x32_bf16(
                false, A.v, false, B.v, (short)0, acc[s], false, false);
        }
        __syncthreads();   // next-tile writes done; this-tile reads done
    }
    // --- store (mean over heads) ---
#pragma unroll
    for (int s = 0; s < 4; ++s) {
        int colG = cbase + wv * 64 + s * 16 + n;
#pragma unroll
        for (int r = 0; r < 8; ++r) {
            int row = i0 + r + 8 * kh;
            out[row * (TT * CC) + colG] = acc[s][r] * (1.0f / HH);
        }
    }
}

// ---------------------------------------------------------------------------
extern "C" void kernel_launch(void* const* d_in, const int* in_sizes, int n_in,
                              void* d_out, int out_size, void* d_ws, size_t ws_size,
                              hipStream_t stream)
{
    const float* x    = (const float*)d_in[0];
    const int*   adj  = (const int*)d_in[1];
    const float* fc_w = (const float*)d_in[2];
    const float* fc_b = (const float*)d_in[3];
    const float* ta_w = (const float*)d_in[4];
    const float* ta_b = (const float*)d_in[5];
    const float* a    = (const float*)d_in[6];
    float* out = (float*)d_out;

    char* ws = (char*)d_ws;
    unsigned short* hbf = (unsigned short*)ws;                 // 64 MiB
    float* src    = (float*)(ws + (size_t)MTOT * KOUT * 2);
    float* dstb   = src    + NN * HH;
    float* rowmax = dstb   + NN * HH;
    float* rowsum = rowmax + NN * HH;

    fc_kernel    <<<dim3((MTOT / 16) * (KOUT / 128)), 256, 0, stream>>>(x, fc_w, fc_b, hbf);
    srcdst_kernel<<<dim3((NN * HH) / 256),            256, 0, stream>>>(hbf, ta_w, ta_b, a, src, dstb);
    stats_kernel <<<dim3(NN),                         256, 0, stream>>>(adj, src, dstb, rowmax, rowsum);
    agg_kernel   <<<dim3(NN / 16, 2),                 256, 0, stream>>>(hbf, adj, src, dstb, rowmax, rowsum, out);
}
